// RationalResNet_26654567039142
// MI455X (gfx1250) — matrix operand order repaired and checked by reading, not compiled, not run on hardware
//
#include <hip/hip_runtime.h>
#include <hip/hip_bf16.h>

typedef __attribute__((ext_vector_type(16))) _Float16 v16h;
typedef __attribute__((ext_vector_type(8)))  _Float16 v8h;
typedef __attribute__((ext_vector_type(8)))  float    v8f;

// ---------------------------------------------------------------------------
// Implicit-GEMM convolution via V_WMMA_F32_16X16X32_F16, fully NHWC.
//   act : NHWC f16, Cin multiple of 32        (B-matrix, K = (kh,kw)-plane x ci)
//   wgt : packed [Cout][KS*KS][Cin] f16       (A-matrix, contiguous ci runs)
//   out : NHWC f32
// 256 threads = 8 waves; wave computes 16(M) x 64(N); block = 32 x 256.
// B=256 and Cout % 32 == 0 -> exact tiling, no M/N guards. Only pad checks.
// ---------------------------------------------------------------------------
template <int KS>
__global__ __launch_bounds__(256) void conv_wmma(
    const _Float16* __restrict__ act, const _Float16* __restrict__ wgt,
    float* __restrict__ out,
    int Cin, int Hin, int Win, int Cout,
    int pad, int stride, int hwShift, int wShift)
{
    constexpr int KK = KS * KS;
    const int lane = threadIdx.x & 31;
    const int wave = threadIdx.x >> 5;
    const int wm   = wave & 1;
    const int wn   = wave >> 1;
    const int half = lane >> 4;
    const int l16  = lane & 15;

    const int mBase = blockIdx.y * 32 + wm * 16;
    const int nBase = blockIdx.x * 256 + wn * 64;
    const int mRow  = mBase + l16;

    // Decode the 4 GEMM columns owned by this lane: n -> (image, ho, wo)
    int ncol[4], bimg[4], ho[4], wo[4];
    #pragma unroll
    for (int t = 0; t < 4; ++t) {
        int n = nBase + t * 16 + l16;
        ncol[t] = n;
        bimg[t] = n >> hwShift;
        int rem = n & ((1 << hwShift) - 1);
        ho[t] = rem >> wShift;
        wo[t] = rem & ((1 << wShift) - 1);
    }

    const _Float16* wRow = wgt + mRow * (KK * Cin) + half * 16;

    v8f acc[4] = {};

    #pragma unroll
    for (int kh = 0; kh < KS; ++kh) {
        #pragma unroll
        for (int kw = 0; kw < KS; ++kw) {
            const int sp = kh * KS + kw;
            const _Float16* wp = wRow + sp * Cin;
            int  bOff[4];
            bool bval[4];
            #pragma unroll
            for (int t = 0; t < 4; ++t) {
                int ih = ho[t] * stride + kh - pad;
                int iw = wo[t] * stride + kw - pad;
                bval[t] = ((unsigned)ih < (unsigned)Hin) &&
                          ((unsigned)iw < (unsigned)Win);
                bOff[t] = (((bimg[t] * Hin + ih) * Win) + iw) * Cin + half * 16;
            }
            for (int ciBase = 0; ciBase < Cin; ciBase += 32) {
                // A fragment: one contiguous 16-half run (2 x b128)
                v8h a0 = *(const v8h*)(wp + ciBase);
                v8h a1 = *(const v8h*)(wp + ciBase + 8);
                v16h a = __builtin_shufflevector(
                    a0, a1, 0, 1, 2, 3, 4, 5, 6, 7,
                    8, 9, 10, 11, 12, 13, 14, 15);
                // 4 B fragments (one 32B load each) + 4 WMMAs reusing A
                #pragma unroll
                for (int t = 0; t < 4; ++t) {
                    v16h bfr = {};
                    if (bval[t])
                        bfr = *(const v16h*)(act + bOff[t] + ciBase);
                    acc[t] = __builtin_amdgcn_wmma_f32_16x16x32_f16(
                        false, a, false, bfr, (short)0, acc[t], false, false);
                }
            }
        }
    }

    // Store: lane owns 8 consecutive output channels per column (2 x float4)
    #pragma unroll
    for (int t = 0; t < 4; ++t) {
        float* op = out + ncol[t] * Cout + mBase + 8 * half;
        *(float4*)(op)     = make_float4(acc[t][0], acc[t][1], acc[t][2], acc[t][3]);
        *(float4*)(op + 4) = make_float4(acc[t][4], acc[t][5], acc[t][6], acc[t][7]);
    }
}

// ---------------------------------------------------------------------------
// Weight repack: OIHW f32 -> [Cout][KK][Cinp] f16 (ci zero-padded to Cinp).
// ---------------------------------------------------------------------------
__global__ __launch_bounds__(256) void pack_w(
    const float* __restrict__ w, _Float16* __restrict__ wp,
    int Cin, int Cinp, int KK, int total)
{
    int i = blockIdx.x * 256 + threadIdx.x;
    if (i >= total) return;
    int ci   = i & (Cinp - 1);
    int rest = i / Cinp;
    int sp   = rest % KK;
    int co   = rest / KK;
    wp[i] = (ci < Cin) ? (_Float16)w[(co * Cin + ci) * KK + sp] : (_Float16)0.f;
}

// ---------------------------------------------------------------------------
// Stem input pack: NCHW f32 (C=3) -> NHWC f16 padded to C=32. B=256, H=W=32.
// ---------------------------------------------------------------------------
__global__ __launch_bounds__(256) void pack_input(
    const float* __restrict__ x, _Float16* __restrict__ out, int total)
{
    int i = blockIdx.x * 256 + threadIdx.x;
    if (i >= total) return;
    int c = i & 31;
    int p = i >> 5;          // (n, h, w) flattened, H=W=32
    int w = p & 31;
    int h = (p >> 5) & 31;
    int n = p >> 10;
    out[i] = (c < 3) ? (_Float16)x[((n * 3 + c) << 10) + (h << 5) + w]
                     : (_Float16)0.f;
}

// ---------------------------------------------------------------------------
// BN batch statistics over NHWC f32: two-stage reduction.
// Stage 1: grid (64 row-blocks, C/64); block = 64 channels x 4 row-lanes.
// Stage 2: per-channel combine -> mean, rsqrt(var+eps).
// ---------------------------------------------------------------------------
__global__ __launch_bounds__(256) void bn_stats_partial(
    const float* __restrict__ x, float* __restrict__ partial, int C, int rows)
{
    const int cl = threadIdx.x & 63;
    const int rl = threadIdx.x >> 6;                   // 0..3
    const int c  = blockIdx.y * 64 + cl;
    const int rowsPerBlock = rows >> 6;                // rows multiple of 64
    const int r0 = blockIdx.x * rowsPerBlock;
    float s = 0.f, s2 = 0.f;
    for (int r = r0 + rl; r < r0 + rowsPerBlock; r += 4) {
        float v = x[r * C + c];
        s += v; s2 += v * v;
    }
    __shared__ float red0[4][64];
    __shared__ float red1[4][64];
    red0[rl][cl] = s;
    red1[rl][cl] = s2;
    __syncthreads();
    if (rl == 0) {
        s  = red0[0][cl] + red0[1][cl] + red0[2][cl] + red0[3][cl];
        s2 = red1[0][cl] + red1[1][cl] + red1[2][cl] + red1[3][cl];
        partial[(blockIdx.x * C + c) * 2]     = s;
        partial[(blockIdx.x * C + c) * 2 + 1] = s2;
    }
}

__global__ __launch_bounds__(256) void bn_stats_final(
    const float* __restrict__ partial, float* __restrict__ stats,
    int C, int rows)
{
    int c = blockIdx.x * 256 + threadIdx.x;
    if (c >= C) return;
    float s = 0.f, s2 = 0.f;
    for (int rb = 0; rb < 64; ++rb) {
        s  += partial[(rb * C + c) * 2];
        s2 += partial[(rb * C + c) * 2 + 1];
    }
    float mean = s / (float)rows;
    float var  = s2 / (float)rows - mean * mean;
    stats[2 * c]     = mean;
    stats[2 * c + 1] = rsqrtf(var + 1e-5f);
}

// ---------------------------------------------------------------------------
// Fused BN-apply (+ optional f16 residual) (+ optional Pade rational),
// NHWC: 8 consecutive channels per thread.  (c & 3) is compile-time per
// unrolled component, so rational coefficients become uniform scalar loads.
// Reads f32 NHWC, writes f16 NHWC.
// ---------------------------------------------------------------------------
__global__ __launch_bounds__(256) void bn_act(
    const float* __restrict__ x, const float* __restrict__ stats,
    const float* __restrict__ gamma, const float* __restrict__ beta,
    const _Float16* __restrict__ resid,
    const float* __restrict__ ratn, const float* __restrict__ ratd,
    _Float16* __restrict__ out, int cMask, int total8)
{
    int i8 = blockIdx.x * 256 + threadIdx.x;
    if (i8 >= total8) return;
    const int base = i8 * 8;
    const int c0 = base & cMask;                 // multiple of 8

    const float4* xp = (const float4*)(x + base);
    float4 u0 = xp[0], u1 = xp[1];
    float y[8] = {u0.x, u0.y, u0.z, u0.w, u1.x, u1.y, u1.z, u1.w};

    #pragma unroll
    for (int j = 0; j < 8; ++j) {
        int c = c0 + j;
        float sc = stats[2 * c + 1] * gamma[c];
        y[j] = (y[j] - stats[2 * c]) * sc + beta[c];
    }
    if (resid) {
        v8h r = *(const v8h*)(resid + base);
        #pragma unroll
        for (int j = 0; j < 8; ++j) y[j] += (float)r[j];
    }
    if (ratn) {
        #pragma unroll
        for (int j = 0; j < 8; ++j) {
            const int g = j & 3;                 // compile-time group
            float P = ratn[g * 6 + 5];
            P = P * y[j] + ratn[g * 6 + 4];
            P = P * y[j] + ratn[g * 6 + 3];
            P = P * y[j] + ratn[g * 6 + 2];
            P = P * y[j] + ratn[g * 6 + 1];
            P = P * y[j] + ratn[g * 6 + 0];
            float Q = ratd[g * 4 + 3];
            Q = Q * y[j] + ratd[g * 4 + 2];
            Q = Q * y[j] + ratd[g * 4 + 1];
            Q = Q * y[j] + ratd[g * 4 + 0];
            Q = Q * y[j];
            y[j] = P / (1.f + fabsf(Q));
        }
    }
    v8h o;
    #pragma unroll
    for (int j = 0; j < 8; ++j) o[j] = (_Float16)y[j];
    *(v8h*)(out + base) = o;
}

// ---------------------------------------------------------------------------
// GAP over 16 pixels (NHWC f16, C=512) and FC 512 -> 10.
// ---------------------------------------------------------------------------
__global__ __launch_bounds__(256) void gap_kernel(
    const _Float16* __restrict__ x, float* __restrict__ out, int total)
{
    int i = blockIdx.x * 256 + threadIdx.x;
    if (i >= total) return;
    int b = i >> 9;                              // C = 512
    int c = i & 511;
    float s = 0.f;
    #pragma unroll
    for (int p = 0; p < 16; ++p) s += (float)x[((b * 16 + p) << 9) + c];
    out[i] = s * (1.f / 16.f);
}

__global__ __launch_bounds__(256) void fc_kernel(
    const float* __restrict__ x, const float* __restrict__ w,
    const float* __restrict__ b, float* __restrict__ out,
    int B, int C, int NC)
{
    int i = blockIdx.x * 256 + threadIdx.x;
    if (i >= B * NC) return;
    int bi = i / NC, j = i - bi * NC;
    float s = b[j];
    for (int c = 0; c < C; ++c) s += x[bi * C + c] * w[c * NC + j];
    out[i] = s;
}

// ---------------------------------------------------------------------------
// Host orchestration
// ---------------------------------------------------------------------------
extern "C" void kernel_launch(void* const* d_in, const int* in_sizes, int n_in,
                              void* d_out, int out_size, void* d_ws, size_t ws_size,
                              hipStream_t stream) {
    (void)in_sizes; (void)n_in; (void)out_size; (void)ws_size;
    const float* x = (const float*)d_in[0];

    struct Blk {
        const float *bn1_b, *bn1_g, *bn2_b, *bn2_g, *conv1, *conv2;
        const float *rat1_d, *rat1_n, *rat2_d, *rat2_n;
        const float *sc_bn_b, *sc_bn_g, *sc_conv;
    };
    Blk blk[8];
    const bool hasSC[8] = {false, false, true, false, true, false, true, false};
    int pi = 1;
    for (int i = 0; i < 8; ++i) {
        blk[i].bn1_b  = (const float*)d_in[pi++];
        blk[i].bn1_g  = (const float*)d_in[pi++];
        blk[i].bn2_b  = (const float*)d_in[pi++];
        blk[i].bn2_g  = (const float*)d_in[pi++];
        blk[i].conv1  = (const float*)d_in[pi++];
        blk[i].conv2  = (const float*)d_in[pi++];
        blk[i].rat1_d = (const float*)d_in[pi++];
        blk[i].rat1_n = (const float*)d_in[pi++];
        blk[i].rat2_d = (const float*)d_in[pi++];
        blk[i].rat2_n = (const float*)d_in[pi++];
        if (hasSC[i]) {
            blk[i].sc_bn_b = (const float*)d_in[pi++];
            blk[i].sc_bn_g = (const float*)d_in[pi++];
            blk[i].sc_conv = (const float*)d_in[pi++];
        } else {
            blk[i].sc_bn_b = blk[i].sc_bn_g = blk[i].sc_conv = nullptr;
        }
    }
    const float* bn1_b  = (const float*)d_in[pi++];
    const float* bn1_g  = (const float*)d_in[pi++];
    const float* conv1w = (const float*)d_in[pi++];
    const float* fc_b   = (const float*)d_in[pi++];
    const float* fc_w   = (const float*)d_in[pi++];
    const float* rat0_d = (const float*)d_in[pi++];
    const float* rat0_n = (const float*)d_in[pi++];

    // Workspace layout (byte offsets, MB-aligned):
    char* ws = (char*)d_ws;
    float*     stats   = (float*)ws;                          // 4 KB
    float*     partial = (float*)(ws + (1u << 20));           // <= 256 KB
    _Float16*  wpack   = (_Float16*)(ws + (2u << 20));        // <= 4.8 MB
    float*     f32a    = (float*)(ws + (16u << 20));          // conv out, 64 MB
    float*     f32b    = (float*)(ws + (80u << 20));          // sc conv out
    _Float16*  h1      = (_Float16*)(ws + (144u << 20));      // f16 act
    _Float16*  h2      = (_Float16*)(ws + (180u << 20));
    _Float16*  h3      = (_Float16*)(ws + (216u << 20));

    const int B = 256;
    auto log2i = [](int v) { int s = 0; while ((1 << s) < v) ++s; return s; };

    auto packw = [&](const float* w, int Cin, int Cinp, int Cout, int KK) {
        int total = Cout * KK * Cinp;
        pack_w<<<dim3((total + 255) / 256), dim3(256), 0, stream>>>(
            w, wpack, Cin, Cinp, KK, total);
    };
    auto conv = [&](const _Float16* in, float* out,
                    int Cinp, int Hin, int Cout, int Hout,
                    int ks, int pad, int stride) {
        int wShift = log2i(Hout);
        int hwShift = 2 * wShift;
        int Ngemm = B << hwShift;                       // multiple of 256
        dim3 grid((unsigned)(Ngemm / 256), (unsigned)(Cout / 32));
        if (ks == 3)
            conv_wmma<3><<<grid, dim3(256), 0, stream>>>(
                in, wpack, out, Cinp, Hin, Hin, Cout, pad, stride,
                hwShift, wShift);
        else
            conv_wmma<1><<<grid, dim3(256), 0, stream>>>(
                in, wpack, out, Cinp, Hin, Hin, Cout, pad, stride,
                hwShift, wShift);
    };
    auto stats_k = [&](const float* xin, int C, int Hout) {
        int rows = B << (2 * log2i(Hout));
        bn_stats_partial<<<dim3(64, (unsigned)(C / 64 ? C / 64 : 1)),
                           dim3(256), 0, stream>>>(xin, partial, C, rows);
        bn_stats_final<<<dim3((unsigned)((C + 255) / 256)), dim3(256), 0,
                         stream>>>(partial, stats, C, rows);
    };
    auto bnact = [&](const float* xin, const float* g, const float* bb,
                     const _Float16* res, const float* rn, const float* rd,
                     _Float16* out, int C, int Hout) {
        int total8 = (B << (2 * log2i(Hout))) * C / 8;
        bn_act<<<dim3((unsigned)((total8 + 255) / 256)), dim3(256), 0, stream>>>(
            xin, stats, g, bb, res, rn, rd, out, C - 1, total8);
    };

    // ---- Stem: pack input (3->32 ch NHWC f16), conv 32->64, BN, rational ----
    {
        int total = B * 32 * 32 * 32;
        pack_input<<<dim3((total + 255) / 256), dim3(256), 0, stream>>>(
            x, h1, total);
    }
    packw(conv1w, 3, 32, 64, 9);
    conv(h1, f32a, 32, 32, 64, 32, 3, 1, 1);
    stats_k(f32a, 64, 32);
    bnact(f32a, bn1_g, bn1_b, nullptr, rat0_n, rat0_d, h2, 64, 32);

    const _Float16* act = h2;
    int cin = 64, hin = 32;
    const int cfgW[8] = {64, 64, 128, 128, 256, 256, 512, 512};
    const int cfgS[8] = {1, 1, 2, 1, 2, 1, 2, 1};
    for (int i = 0; i < 8; ++i) {
        const int w = cfgW[i], s = cfgS[i];
        const int hout = hin / s;

        // main: conv1 -> BN1 -> rational1
        packw(blk[i].conv1, cin, cin, w, 9);
        conv(act, f32a, cin, hin, w, hout, 3, 1, s);
        stats_k(f32a, w, hout);
        bnact(f32a, blk[i].bn1_g, blk[i].bn1_b, nullptr,
              blk[i].rat1_n, blk[i].rat1_d, h1, w, hout);

        // main: conv2
        packw(blk[i].conv2, w, w, w, 9);
        conv(h1, f32a, w, hout, w, hout, 3, 1, 1);

        // shortcut
        const _Float16* res;
        if (blk[i].sc_conv) {
            packw(blk[i].sc_conv, cin, cin, w, 1);
            conv(act, f32b, cin, hin, w, hout, 1, 0, s);
            stats_k(f32b, w, hout);
            bnact(f32b, blk[i].sc_bn_g, blk[i].sc_bn_b, nullptr,
                  nullptr, nullptr, h3, w, hout);
            res = h3;
        } else {
            res = act;
        }

        // BN2 + residual + rational2
        stats_k(f32a, w, hout);
        bnact(f32a, blk[i].bn2_g, blk[i].bn2_b, res,
              blk[i].rat2_n, blk[i].rat2_d, h2, w, hout);
        act = h2;
        cin = w;
        hin = hout;
    }

    // ---- GAP (4x4 -> 1) + FC (512 -> 10) ----
    float* pooled = f32b;
    gap_kernel<<<dim3((B * 512 + 255) / 256), dim3(256), 0, stream>>>(
        act, pooled, B * 512);
    fc_kernel<<<dim3((B * 10 + 255) / 256), dim3(256), 0, stream>>>(
        pooled, fc_w, fc_b, (float*)d_out, B, 512, 10);
}
